// EventSNNFlowNetLiteV2_88742614270102
// MI455X (gfx1250) — matrix operand order repaired and checked by reading, not compiled, not run on hardware
//
#include <hip/hip_runtime.h>
#include <hip/hip_bf16.h>

// ---------------------------------------------------------------------------
// EventSNNFlowNetLite on MI455X (gfx1250): implicit-GEMM convs on
// v_wmma_f32_16x16x32_bf16. Per-wave 16x(NFRAG*16) tiles so each im2col A
// fragment feeds NFRAG WMMAs; double-buffered LDS weight slabs; fused LIF.
// ---------------------------------------------------------------------------

typedef __attribute__((ext_vector_type(16))) __bf16 v16bf;
typedef __attribute__((ext_vector_type(8)))  float  v8f;

static __device__ __forceinline__ unsigned short f2bfu(float f) {
  union { float f; unsigned u; } v; v.f = f;
  unsigned u = v.u;
  unsigned r = u + 0x7FFFu + ((u >> 16) & 1u);  // round-to-nearest-even
  return (unsigned short)(r >> 16);
}
static __device__ __forceinline__ __bf16 u2bf(unsigned short s) {
  union { unsigned short s; __bf16 h; } c; c.s = s; return c.h;
}
// 16-bit A-matrix element->K map (ISA 7.12.2): element e of lane-group grp
// holds K = a_k_of(e,grp); lane&15 is the M row.
static __device__ __forceinline__ int a_k_of(int e, int grp) {
  int v = e >> 1, p = e & 1;
  int base = (v < 4) ? 0 : 16;
  int vv   = (v < 4) ? v : v - 4;
  return base + grp * 8 + vv * 2 + p;
}

// Implicit-GEMM conv: M = Nb*outH*outW, N = Cout, K = Cin*KSZ*KSZ (pad 32).
// Wave tile: 16 pixels x NFRAG*16 channels. 4 waves/block share the LDS
// weight slab (all waves use the same N range, distinct M tiles).
template <int KSZ, int STRIDE, int PAD, int UPS, int NFRAG>
__global__ __launch_bounds__(128)
void conv_wmma_lif(const float* __restrict__ in, long long inBS,
                   const unsigned short* __restrict__ wB,
                   const float* __restrict__ bias,
                   float* __restrict__ out, float* __restrict__ mem,
                   int Nb, int Cout, int srcH, int srcW,
                   int outH, int outW, int Kact, int Kpad,
                   int doLif, float scale, int accum) {
  constexpr int NF16 = NFRAG * 16;
  constexpr int KK   = KSZ * KSZ;
  __shared__ unsigned short sB[2][32 * NF16];   // double-buffered K-slabs

  const int tid  = threadIdx.x;
  const int wave = tid >> 5;
  const int lane = tid & 31;
  const int grp  = lane >> 4;
  const int lnm  = lane & 15;

  const int mTiles = (Nb * outH * outW) >> 4;
  const int myTile = blockIdx.x * 4 + wave;
  const bool active = (myTile < mTiles);
  const int m0 = (active ? myTile : (mTiles - 1)) << 4;
  const int n0 = blockIdx.y * NF16;

  const int inH = UPS ? (srcH << 1) : srcH;
  const int inW = UPS ? (srcW << 1) : srcW;
  const int hw  = outH * outW;

  // pixel this lane's A row covers (M = lnm)
  const int gm  = m0 + lnm;
  const int nI  = gm / hw;
  const int rem = gm - nI * hw;
  const int oy  = rem / outW;
  const int ox  = rem - oy * outW;
  const int iy0 = oy * STRIDE - PAD;
  const int ix0 = ox * STRIDE - PAD;
  const float* __restrict__ inN = in + (long long)nI * inBS;

  v8f acc[NFRAG];
  #pragma unroll
  for (int f = 0; f < NFRAG; ++f)
    #pragma unroll
    for (int i = 0; i < 8; ++i) acc[f][i] = 0.0f;

  // cooperative stage of one 32 x NF16 weight slab (bf16) into LDS buffer
  auto stage = [&](int ks, int buf) {
    const int kb = ks << 5;
    constexpr int STG = (32 * NF16) / 128;      // entries per thread
    #pragma unroll
    for (int i = 0; i < STG; ++i) {
      const int idx = tid + i * 128;
      const int kl  = idx / NF16;
      const int n   = idx - kl * NF16;
      const int kgl = kb + kl;
      const int ch  = n0 + n;
      unsigned short v = 0;
      if (kgl < Kact && ch < Cout) v = wB[(long long)kgl * Cout + ch];
      sB[buf][kl * NF16 + n] = v;
    }
    if (kb + 64 < Kact)                          // slab after next -> L2/L0
      __builtin_prefetch(&wB[(long long)(kb + 64) * Cout + n0], 0, 1);
  };

  const int ksteps = Kpad >> 5;
  stage(0, 0);
  for (int ks = 0; ks < ksteps; ++ks) {
    __syncthreads();
    const int buf = ks & 1;
    const int kb  = ks << 5;

    // ---- A fragment: on-the-fly im2col gather (constants folded) ----------
    v16bf af;
    #pragma unroll
    for (int e = 0; e < 16; ++e) {
      const int k = kb + a_k_of(e, grp);
      float fv = 0.0f;
      if (k < Kact) {
        const int ci = k / KK;                   // KK compile-time
        const int r2 = k - ci * KK;
        const int ky = r2 / KSZ;
        const int kx = r2 - ky * KSZ;
        const int iy = iy0 + ky;
        const int ix = ix0 + kx;
        if (iy >= 0 && iy < inH && ix >= 0 && ix < inW) {
          const int sy = UPS ? (iy >> 1) : iy;
          const int sx = UPS ? (ix >> 1) : ix;
          fv = inN[((long long)ci * srcH + sy) * srcW + sx];
        }
      }
      af[e] = u2bf(f2bfu(fv));
    }

    if (ks + 1 < ksteps) stage(ks + 1, buf ^ 1);

    // ---- NFRAG B fragments (strided 16-bit LDS reads) + WMMAs ---------------------
    const unsigned short* bcol = &sB[buf][lnm];  // N column this lane supplies
    #pragma unroll
    for (int f = 0; f < NFRAG; ++f) {
      union { v16bf v; unsigned short u[16]; } bu;
      #pragma unroll
      for (int s = 0; s < 16; ++s)
        bu.u[s] = bcol[a_k_of(s, grp) * NF16 + f * 16];
      acc[f] = __builtin_amdgcn_wmma_f32_16x16x32_bf16(
          false, af, false, bu.v, (short)0, acc[f], false, false);
    }
  }

  // ---- epilogue: bias + (LIF | scale/accumulate) --------------------------
  if (active) {
    #pragma unroll
    for (int r = 0; r < 8; ++r) {
      const int gm2 = m0 + r + (grp << 3);       // C/D: VGPR r -> M = r (+8)
      const int nI2 = gm2 / hw;
      const int rm  = gm2 - nI2 * hw;
      const long long pixOff =
          ((long long)nI2 * Cout * outH + (rm / outW)) * outW + (rm % outW);
      #pragma unroll
      for (int f = 0; f < NFRAG; ++f) {
        const int ch = n0 + f * 16 + lnm;
        if (ch < Cout) {
          const long long off = pixOff + (long long)ch * hw;
          float v = acc[f][r] + (bias ? bias[ch] : 0.0f);
          if (doLif) {
            float m = mem[off];
            m += (v - m) * 0.5f;                 // 1/TAU
            const float s = (m - 1.0f) > 0.0f ? 1.0f : 0.0f;
            mem[off] = m - s;                    // soft reset
            out[off] = s;
          } else {
            v *= scale;
            if (accum) out[off] += v; else out[off] = v;
          }
        }
      }
    }
  }
}

// Running mean: mn += (s - mn) * inv  (sizes divisible by 4 -> float4)
__global__ void mean_update(float4* __restrict__ mn, const float4* __restrict__ s,
                            float inv, long long n4) {
  long long i = (long long)blockIdx.x * blockDim.x + threadIdx.x;
  if (i < n4) {
    float4 m = mn[i], v = s[i];
    m.x += (v.x - m.x) * inv;  m.y += (v.y - m.y) * inv;
    m.z += (v.z - m.z) * inv;  m.w += (v.w - m.w) * inv;
    mn[i] = m;
  }
}

// OIHW f32 -> [Kpad][Cout] bf16 (K = ci*KH*KW + ky*KW + kx, zero-padded)
__global__ void convert_w(const float* __restrict__ w, unsigned short* __restrict__ o,
                          int Cout, int Kact, int Kpad) {
  int idx = blockIdx.x * blockDim.x + threadIdx.x;
  if (idx >= Kpad * Cout) return;
  int k = idx / Cout, co = idx - k * Cout;
  float f = (k < Kact) ? w[(long long)co * Kact + k] : 0.0f;
  o[idx] = f2bfu(f);
}

// ---------------------------------------------------------------------------
template <int KSZ, int STRIDE, int PAD, int UPS, int NFRAG>
static void launch_conv(hipStream_t stream, const float* in, long long inBS,
                        const unsigned short* wb, const float* bias,
                        float* out, float* mem, int Nb, int Cin, int Cout,
                        int srcH, int srcW, int oH, int oW,
                        int doLif, float scale, int accum) {
  const int Kact = Cin * KSZ * KSZ, Kp = (Kact + 31) & ~31;
  const int mTiles = (Nb * oH * oW) >> 4;
  dim3 g((mTiles + 3) / 4, (Cout + NFRAG * 16 - 1) / (NFRAG * 16)), b(128);
  conv_wmma_lif<KSZ, STRIDE, PAD, UPS, NFRAG><<<g, b, 0, stream>>>(
      in, inBS, wb, bias, out, mem, Nb, Cout, srcH, srcW, oH, oW,
      Kact, Kp, doLif, scale, accum);
}

extern "C" void kernel_launch(void* const* d_in, const int* in_sizes, int n_in,
                              void* d_out, int out_size, void* d_ws, size_t ws_size,
                              hipStream_t stream) {
  (void)in_sizes; (void)n_in; (void)out_size; (void)ws_size;
  const float* x     = (const float*)d_in[0];
  const float* w_e1  = (const float*)d_in[1];  const float* b_e1 = (const float*)d_in[2];
  const float* w_e2  = (const float*)d_in[3];  const float* b_e2 = (const float*)d_in[4];
  const float* w_e3  = (const float*)d_in[5];  const float* b_e3 = (const float*)d_in[6];
  const float* w_d3  = (const float*)d_in[7];  const float* b_d3 = (const float*)d_in[8];
  const float* w_d2  = (const float*)d_in[9];  const float* b_d2 = (const float*)d_in[10];
  const float* w_d1  = (const float*)d_in[11]; const float* b_d1 = (const float*)d_in[12];
  const float* w_sk2 = (const float*)d_in[13];
  const float* w_sk1 = (const float*)d_in[14];
  const float* w_fl  = (const float*)d_in[15]; const float* b_fl = (const float*)d_in[16];

  const int Nb = 8, T = 10, H = 256, W = 256, Bc = 32;

  // --- workspace layout ----------------------------------------------------
  size_t off = 0;
  char* ws = (char*)d_ws;
  auto allocF = [&](size_t n) {
    float* p = (float*)(ws + off);
    off = (off + n * sizeof(float) + 255) & ~(size_t)255;
    return p;
  };
  auto allocU = [&](size_t n) {
    unsigned short* p = (unsigned short*)(ws + off);
    off = (off + n * sizeof(unsigned short) + 255) & ~(size_t)255;
    return p;
  };
  const size_t s1n = (size_t)Nb * Bc   * 128 * 128;
  const size_t s2n = (size_t)Nb * 2*Bc *  64 *  64;
  const size_t s3n = (size_t)Nb * 4*Bc *  32 *  32;
  const size_t d1n = (size_t)Nb * Bc   * 256 * 256;
  float* s1  = allocF(s1n);
  float* s2  = allocF(s2n);
  float* s3  = allocF(s3n);
  float* d3b = allocF(s2n);
  float* d2b = allocF(s1n);
  float* d1b = allocF(d1n);
  size_t stateOff = off;                   // recurrent state, zeroed per call
  float* me1 = allocF(s1n);  float* me2 = allocF(s2n);  float* me3 = allocF(s3n);
  float* md3 = allocF(s2n);  float* md2 = allocF(s1n);  float* md1 = allocF(d1n);
  float* mn1 = allocF(s1n);  float* mn2 = allocF(s2n);
  size_t stateEnd = off;
  auto kpad = [](int cin, int k) { return ((cin * k * k) + 31) & ~31; };
  unsigned short* we1 = allocU((size_t)kpad(2,    5) * Bc);
  unsigned short* we2 = allocU((size_t)kpad(Bc,   3) * 2*Bc);
  unsigned short* we3 = allocU((size_t)kpad(2*Bc, 3) * 4*Bc);
  unsigned short* wd3 = allocU((size_t)kpad(4*Bc, 3) * 2*Bc);
  unsigned short* wd2 = allocU((size_t)kpad(2*Bc, 3) * Bc);
  unsigned short* wd1 = allocU((size_t)kpad(Bc,   3) * Bc);
  unsigned short* wk2 = allocU((size_t)kpad(2*Bc, 1) * 2*Bc);
  unsigned short* wk1 = allocU((size_t)kpad(Bc,   1) * Bc);
  unsigned short* wfl = allocU((size_t)kpad(Bc,   3) * 2);

  hipMemsetAsync(ws + stateOff, 0, stateEnd - stateOff, stream);

  // --- weight conversion ---------------------------------------------------
  auto convW = [&](const float* w, unsigned short* dst, int Cout, int Cin, int k) {
    int Kact = Cin * k * k, Kp = (Kact + 31) & ~31;
    int tot = Kp * Cout;
    convert_w<<<(tot + 255) / 256, 256, 0, stream>>>(w, dst, Cout, Kact, Kp);
  };
  convW(w_e1, we1, Bc,   2,    5);
  convW(w_e2, we2, 2*Bc, Bc,   3);
  convW(w_e3, we3, 4*Bc, 2*Bc, 3);
  convW(w_d3, wd3, 2*Bc, 4*Bc, 3);
  convW(w_d2, wd2, Bc,   2*Bc, 3);
  convW(w_d1, wd1, Bc,   Bc,   3);
  convW(w_sk2, wk2, 2*Bc, 2*Bc, 1);
  convW(w_sk1, wk1, Bc,   Bc,   1);
  convW(w_fl,  wfl, 2,    Bc,   3);

  // --- recurrent time loop -------------------------------------------------
  for (int t = 0; t < T; ++t) {
    const float* xt = x + (size_t)t * 2 * H * W;           // [N,T,2,H,W]
    launch_conv<5,2,2,0,2>(stream, xt, (long long)T*2*H*W, we1, b_e1, s1, me1,
                           Nb, 2,    Bc,   256, 256, 128, 128, 1, 1.f, 0);
    launch_conv<3,2,1,0,4>(stream, s1, (long long)Bc*128*128, we2, b_e2, s2, me2,
                           Nb, Bc,   2*Bc, 128, 128,  64,  64, 1, 1.f, 0);
    launch_conv<3,2,1,0,4>(stream, s2, (long long)2*Bc*64*64, we3, b_e3, s3, me3,
                           Nb, 2*Bc, 4*Bc,  64,  64,  32,  32, 1, 1.f, 0);

    const float inv = 1.0f / (float)(t + 1);
    mean_update<<<(int)((s1n/4 + 255) / 256), 256, 0, stream>>>(
        (float4*)mn1, (const float4*)s1, inv, (long long)(s1n/4));
    mean_update<<<(int)((s2n/4 + 255) / 256), 256, 0, stream>>>(
        (float4*)mn2, (const float4*)s2, inv, (long long)(s2n/4));

    launch_conv<3,1,1,1,4>(stream, s3,  (long long)4*Bc*32*32, wd3, b_d3, d3b, md3,
                           Nb, 4*Bc, 2*Bc,  32,  32,  64,  64, 1, 1.f, 0);
    launch_conv<1,1,0,0,4>(stream, mn2, (long long)2*Bc*64*64, wk2, nullptr, d3b, nullptr,
                           Nb, 2*Bc, 2*Bc,  64,  64,  64,  64, 0, 1.f, 1);
    launch_conv<3,1,1,1,2>(stream, d3b, (long long)2*Bc*64*64, wd2, b_d2, d2b, md2,
                           Nb, 2*Bc, Bc,    64,  64, 128, 128, 1, 1.f, 0);
    launch_conv<1,1,0,0,2>(stream, mn1, (long long)Bc*128*128, wk1, nullptr, d2b, nullptr,
                           Nb, Bc,   Bc,   128, 128, 128, 128, 0, 1.f, 1);
    launch_conv<3,1,1,1,2>(stream, d2b, (long long)Bc*128*128, wd1, b_d1, d1b, md1,
                           Nb, Bc,   Bc,   128, 128, 256, 256, 1, 1.f, 0);

    if (t == T - 1)     // flow head only matters on the last step (scale 16)
      launch_conv<3,1,1,0,1>(stream, d1b, (long long)Bc*256*256, wfl, b_fl,
                             (float*)d_out, nullptr,
                             Nb, Bc, 2, 256, 256, 256, 256, 0, 16.f, 0);
  }
}
